// HungarianMatcher_60825326846459
// MI455X (gfx1250) — matrix operand order repaired and checked by reading, not compiled, not run on hardware
//
#include <hip/hip_runtime.h>

typedef __attribute__((ext_vector_type(16))) _Float16 v16h;
typedef __attribute__((ext_vector_type(8)))  _Float16 v8h;
typedef __attribute__((ext_vector_type(8)))  float    v8f;

#define BN    64
#define QN    900
#define TN    256
#define CN    256
#define QT    57          // ceil(900/16) q-tiles
#define NCHK  15          // ceil(57/4) q-tile chunks, 4 tiles per block
#define NJT   4           // q-tiles per block
#define ESTR  264         // f16 row stride: 528B, 16B-aligned, staggered banks

__device__ __forceinline__ float fast_rcp(float x) {
    return __builtin_amdgcn_rcpf(x);   // v_rcp_f32: ~1ulp, inf/NaN propagate
}

__device__ __forceinline__ float nan_to_num_big(float r) {
    // nan -> 1e6, +inf -> 1e6, -inf -> -1e6 (finite costs are << 1e6)
    if (r != r) return 1.0e6f;
    return fminf(fmaxf(r, -1.0e6f), 1.0e6f);
}

__global__ __launch_bounds__(256) void matcher_cost_kernel(
    const float* __restrict__ logits,   // [B,Q,C]
    const float* __restrict__ pboxes,   // [B,Q,4] cxcywh
    const int*   __restrict__ labels,   // [B,T]
    const float* __restrict__ tboxes,   // [B,T,4] cxcywh
    float*       __restrict__ out)      // [B,Q,T]
{
    __shared__ __align__(16) _Float16 smE[NJT * 16 * ESTR]; // exp(logit-rowmax) f16
    __shared__ float red[16][17];
    __shared__ float invrow[NJT * 16];
    __shared__ __align__(16) float pbox[NJT * 16 * 4];

    const int tid   = threadIdx.x;
    const int b     = blockIdx.y;
    const int chunk = blockIdx.x;       // 0..14; q-tiles {chunk + 15*j}

    // ---- pred boxes for the 4 q-tiles into LDS (64 rows x 4 floats) ----
    {
        int rr = tid >> 2, cc = tid & 3;            // rr 0..63
        int jj = rr >> 4;
        int q  = min((chunk + NCHK * jj) * 16 + (rr & 15), QN - 1);
        pbox[tid] = pboxes[((size_t)b * QN + q) * 4 + cc];
    }

    // ---- row softmax for 4 q-tiles (16 rows each), exp stored as packed f16 ----
    const int r = tid >> 4;            // 0..15 row within a tile
    const int c = tid & 15;            // 16-wide chunk of C
    for (int j = 0; j < NJT; ++j) {
        const int qt = chunk + NCHK * j;
        const int qc = min(qt * 16 + r, QN - 1);
        const float* lrow = logits + ((size_t)b * QN + qc) * CN + c * 16;
        float l[16];
        #pragma unroll
        for (int v = 0; v < 4; ++v) {
            float4 f = *reinterpret_cast<const float4*>(lrow + v * 4);
            l[4*v+0] = f.x; l[4*v+1] = f.y; l[4*v+2] = f.z; l[4*v+3] = f.w;
        }
        float lm = l[0];
        #pragma unroll
        for (int i = 1; i < 16; ++i) lm = fmaxf(lm, l[i]);
        red[r][c] = lm;
        __syncthreads();
        float rowmax = red[r][0];
        #pragma unroll
        for (int v = 1; v < 16; ++v) rowmax = fmaxf(rowmax, red[r][v]);
        __syncthreads();

        float s = 0.0f;
        v8h e0, e1;
        #pragma unroll
        for (int i = 0; i < 8; ++i) {
            float e = __expf(l[i] - rowmax);
            s += e; e0[i] = (_Float16)e;
        }
        #pragma unroll
        for (int i = 0; i < 8; ++i) {
            float e = __expf(l[8 + i] - rowmax);
            s += e; e1[i] = (_Float16)e;
        }
        _Float16* erow = smE + (j * 16 + r) * ESTR + c * 16;   // 32B-offset, 16B-aligned
        *reinterpret_cast<v8h*>(erow)     = e0;                // 2x ds_store_b128
        *reinterpret_cast<v8h*>(erow + 8) = e1;

        red[r][c] = s;
        __syncthreads();
        float rs = 0.0f;
        #pragma unroll
        for (int v = 0; v < 16; ++v) rs += red[r][v];
        if (c == 0) invrow[j * 16 + r] = 1.0f / rs;
        __syncthreads();
    }

    // ---- K-outer one-hot GEMM: B fragments built once per K-step,
    //      consumed by 8 WMMAs (4 q-tiles x 2 t-tiles) ----
    const int lane   = tid & 31;
    const int waveId = tid >> 5;        // 0..7
    const int wt0    = waveId * 32;     // T columns [wt0, wt0+32)
    const int n      = lane & 15;
    const int hiHalf = lane >> 4;

    const int lab0 = labels[b * TN + wt0 + n];
    const int lab1 = labels[b * TN + wt0 + 16 + n];
    // B k order chosen here (contiguous, A matches): lanes 0-15 hold K=kb..kb+15, lanes 16-31 K=kb+16..kb+31.
    const int base0 = lab0 - (hiHalf << 4);
    const int base1 = lab1 - (hiHalf << 4);

    v8f acc[NJT][2];
    #pragma unroll
    for (int j = 0; j < NJT; ++j) { acc[j][0] = (v8f){}; acc[j][1] = (v8f){}; }

    const int khbA = hiHalf * 16;       // A layout K sub-offset
    const _Float16* Abase = smE + n * ESTR + khbA;

    #pragma unroll
    for (int kb = 0; kb < 8; ++kb) {
        const int Kbase = kb * 32;
        const int d0 = base0 - Kbase;   // one-hot position within this fragment
        const int d1 = base1 - Kbase;
        v16h bf0, bf1;
        #pragma unroll
        for (int e = 0; e < 16; ++e) {  // inline-constant compares
            bf0[e] = (d0 == e) ? (_Float16)1.0f : (_Float16)0.0f;
            bf1[e] = (d1 == e) ? (_Float16)1.0f : (_Float16)0.0f;
        }
        #pragma unroll
        for (int j = 0; j < NJT; ++j) {
            const _Float16* Arow = Abase + (j * 16) * ESTR + Kbase;
            v8h lo = *reinterpret_cast<const v8h*>(Arow);
            v8h hi = *reinterpret_cast<const v8h*>(Arow + 8);
            v16h af = __builtin_shufflevector(lo, hi,
                        0,1,2,3,4,5,6,7,8,9,10,11,12,13,14,15);
            acc[j][0] = __builtin_amdgcn_wmma_f32_16x16x32_f16(
                            false, af, false, bf0, (short)0, acc[j][0], false, false);
            acc[j][1] = __builtin_amdgcn_wmma_f32_16x16x32_f16(
                            false, af, false, bf1, (short)0, acc[j][1], false, false);
        }
    }

    // ---- fused epilogue: bbox L1 + GIoU + combine + nan_to_num ----
    float4 tb[2];
    tb[0] = *reinterpret_cast<const float4*>(tboxes + ((size_t)b * TN + wt0 + n) * 4);
    tb[1] = *reinterpret_cast<const float4*>(tboxes + ((size_t)b * TN + wt0 + 16 + n) * 4);

    #pragma unroll
    for (int j = 0; j < NJT; ++j) {
        const int qt = chunk + NCHK * j;
        // hoist the 8 pred-box reads for this q-tile out of the 2-tile loop
        float4 pbj[8];
        #pragma unroll
        for (int i = 0; i < 8; ++i) {
            const int m = i + 8 * hiHalf;
            pbj[i] = *reinterpret_cast<const float4*>(pbox + (j * 16 + m) * 4);
        }
        #pragma unroll
        for (int tile = 0; tile < 2; ++tile) {
            const v8f a = acc[j][tile];
            const int t = wt0 + tile * 16 + n;
            const float4 T4 = tb[tile];
            const float tx0 = T4.x - 0.5f * T4.z, ty0 = T4.y - 0.5f * T4.w;
            const float tx1 = T4.x + 0.5f * T4.z, ty1 = T4.y + 0.5f * T4.w;
            const float ta  = T4.z * T4.w;

            #pragma unroll
            for (int i = 0; i < 8; ++i) {
                const int m = i + 8 * hiHalf;       // C/D layout row
                const int q = qt * 16 + m;
                const float4 pb = pbj[i];
                const float px0 = pb.x - 0.5f * pb.z, py0 = pb.y - 0.5f * pb.w;
                const float px1 = pb.x + 0.5f * pb.z, py1 = pb.y + 0.5f * pb.w;
                const float pa  = pb.z * pb.w;

                const float l1 = fabsf(pb.x - T4.x) + fabsf(pb.y - T4.y) +
                                 fabsf(pb.z - T4.z) + fabsf(pb.w - T4.w);

                const float iw = fmaxf(fminf(px1, tx1) - fmaxf(px0, tx0), 0.0f);
                const float ih = fmaxf(fminf(py1, ty1) - fmaxf(py0, ty0), 0.0f);
                const float inter = iw * ih;
                const float uni   = pa + ta - inter;
                const float iou   = inter * fast_rcp(uni);       // v_rcp+v_mul
                const float cw = fmaxf(fmaxf(px1, tx1) - fminf(px0, tx0), 0.0f);
                const float ch = fmaxf(fmaxf(py1, ty1) - fminf(py0, ty0), 0.0f);
                const float ca = cw * ch;
                const float giou = iou - (ca - uni) * fast_rcp(ca);

                const float cost_class = -a[i] * invrow[j * 16 + m];
                float rcost = cost_class + 5.0f * l1 - 2.0f * giou;
                rcost = nan_to_num_big(rcost);

                if (q < QN)
                    out[((size_t)b * QN + q) * TN + t] = rcost;
            }
        }
    }
}

extern "C" void kernel_launch(void* const* d_in, const int* in_sizes, int n_in,
                              void* d_out, int out_size, void* d_ws, size_t ws_size,
                              hipStream_t stream) {
    const float* logits = (const float*)d_in[0];   // [64,900,256] f32
    const float* pboxes = (const float*)d_in[1];   // [64,900,4]   f32
    const int*   labels = (const int*)  d_in[2];   // [64,256]     i32
    const float* tboxes = (const float*)d_in[3];   // [64,256,4]   f32
    float* out = (float*)d_out;                    // [64,900,256] f32

    dim3 grid(NCHK, BN);   // 15 q-chunks (4 tiles each, interleaved) x 64 batches
    dim3 block(256);       // 8 wave32s; each wave owns two 16-wide T-tiles
    matcher_cost_kernel<<<grid, block, 0, stream>>>(logits, pboxes, labels, tboxes, out);
}
